// NextGram_5927054868513
// MI455X (gfx1250) — matrix operand order repaired and checked by reading, not compiled, not run on hardware
//
#include <hip/hip_runtime.h>

typedef unsigned int u32;
typedef unsigned short u16;
typedef unsigned long long u64;
typedef __attribute__((ext_vector_type(16))) __bf16 v16bf;
typedef __attribute__((ext_vector_type(8)))  float  v8f;
typedef __attribute__((ext_vector_type(4)))  u32    v4u;
typedef __attribute__((ext_vector_type(8)))  int    v8i;
typedef __attribute__((ext_vector_type(4)))  int    v4i;

union FragU { uint4 q[2]; v16bf v; };

// Problem sizes
#define VSZ 16000
#define BSZ 32
#define TSZ 256
#define ESZ 512
#define G4  2048
#define BT  8192   // B*T

#if defined(__HIP_DEVICE_COMPILE__) && __has_builtin(__builtin_amdgcn_tensor_load_to_lds) && __has_builtin(__builtin_amdgcn_s_wait_tensorcnt)
#define USE_TDM 1
#else
#define USE_TDM 0
#endif

// Workspace layout (bytes)
static const size_t OFF_X    = 0;                       // X bf16 [8192,512]  (8 MB)
static const size_t OFF_G0X  = OFF_X    + 8388608;      // G0x f32 [8192,2048] (64 MB)
static const size_t OFF_WX0P = OFF_G0X  + 67108864;     // packed bf16 weights
static const size_t OFF_WH0P = OFF_WX0P + 2097152;
static const size_t OFF_WX1P = OFF_WH0P + 2097152;
static const size_t OFF_WH1P = OFF_WX1P + 2097152;
static const size_t OFF_WDP  = OFF_WH1P + 2097152;      // Wd packed (16 MB)
static const size_t OFF_H1   = OFF_WDP  + 16384000;     // H1 bf16 [8192,512] batch-major (8 MB)
static const size_t OFF_H0S  = OFF_H1   + 8388608;      // h0 double buffer 2*[32,512] bf16
static const size_t OFF_H1S  = OFF_H0S  + 65536;        // h1 double buffer
static const size_t OFF_CNT  = OFF_H1S  + 65536;        // barrier counter
static const size_t OFF_NLL  = OFF_CNT  + 256;          // nll accumulator

__device__ __forceinline__ u16 f2bf(float f) {
    u32 u = __float_as_uint(f);
    u32 r = (u + 0x7FFFu + ((u >> 16) & 1u)) >> 16;   // RNE
    return (u16)r;
}
__device__ __forceinline__ float sigmoidf_(float x) { return 1.0f / (1.0f + __expf(-x)); }

// One 16x16 output tile: acc += A[16 x K] * Bpacked[:, nt].  A bf16 row-major (lda elems),
// Bp pre-packed in WMMA B-fragment order: [(nt*KT + kt)*256 + lane*8 + v] dwords.
__device__ __forceinline__ v8f tile_mm(const u16* __restrict__ A, int lda,
                                       const u32* __restrict__ Bp, int KT, int nt,
                                       v8f acc, int lane) {
    const int g = lane >> 4, ln = lane & 15;
    const u32* bbase = Bp + ((size_t)nt * KT) * 256 + lane * 8;
#pragma unroll 4
    for (int kt = 0; kt < KT; ++kt) {
        FragU a, b;
        const uint4* ap = (const uint4*)(A + (size_t)ln * lda + kt * 32);
        a.q[0] = ap[g];
        a.q[1] = ap[2 + g];
        const uint4* bq = (const uint4*)(bbase + (size_t)kt * 256);
        b.q[0] = bq[0];
        b.q[1] = bq[1];
        acc = __builtin_amdgcn_wmma_f32_16x16x32_bf16(false, a.v, false, b.v,
                                                      (short)0, acc, false, false);
    }
    return acc;
}

// ---------- embedding gather (fp32 -> bf16, time-major rows t*B+b) ----------
__global__ __launch_bounds__(256) void embed_kernel(const int* __restrict__ inputs,
                                                    const float* __restrict__ Em,
                                                    u16* __restrict__ X) {
    size_t id = (size_t)blockIdx.x * 256 + threadIdx.x;
    if (id >= (size_t)BT * ESZ) return;
    int e = (int)(id & (ESZ - 1));
    int row = (int)(id >> 9);
    int t = row >> 5, b = row & 31;
    int tok = inputs[b * TSZ + t];
    X[id] = f2bf(Em[(size_t)tok * ESZ + e]);
}

// ---------- pack fp32 [K,N] into bf16 WMMA B fragments ----------
__global__ __launch_bounds__(256) void pack_b_kernel(const float* __restrict__ B,
                                                     u32* __restrict__ out, int K, int N) {
    int KT = K >> 5, NT = N >> 4;
    size_t ndw = (size_t)KT * NT * 256;
    size_t id = (size_t)blockIdx.x * 256 + threadIdx.x;
    if (id >= ndw) return;
    int v = (int)(id & 7);
    int lane = (int)((id >> 3) & 31);
    size_t tile = id >> 8;
    int kt = (int)(tile % KT);
    int nt = (int)(tile / KT);
    int g = lane >> 4, ln = lane & 15;
    int n = nt * 16 + ln;
    int k0 = kt * 32 + (v >> 2) * 16 + g * 8 + 2 * (v & 3);
    u32 lo = f2bf(B[(size_t)k0 * N + n]);
    u32 hi = f2bf(B[(size_t)(k0 + 1) * N + n]);
    out[id] = lo | (hi << 16);
}

// ---------- bf16 WMMA GEMM, 4Mx2N register blocking per wave ----------
// C = A(bf16)[M,K] @ Bp + bias (fp32 out).  8 waves/block share the N-pair -> B frag broadcast.
__global__ __launch_bounds__(256) void gemm_bf16_kernel(const u16* __restrict__ A,
                                                        const u32* __restrict__ Bp,
                                                        const float* __restrict__ bias,
                                                        float* __restrict__ C,
                                                        int M, int N, int K) {
    const int TM4 = M >> 6, TN2 = N >> 5, KT = K >> 5;
    const int wave = threadIdx.x >> 5, lane = threadIdx.x & 31;
    long grp = (long)blockIdx.x * 8 + wave;
    if (grp >= (long)TM4 * TN2) return;
    const int ng = (int)(grp / TM4);
    const int mg = (int)(grp % TM4);
    const int nt0 = ng * 2;
    const int g = lane >> 4, ln = lane & 15;

    v8f acc[4][2];
    {
        float bv0 = bias ? bias[nt0 * 16 + ln] : 0.0f;
        float bv1 = bias ? bias[nt0 * 16 + 16 + ln] : 0.0f;
#pragma unroll
        for (int m = 0; m < 4; ++m)
#pragma unroll
            for (int r = 0; r < 8; ++r) { acc[m][0][r] = bv0; acc[m][1][r] = bv1; }
    }
    const u32* b0p = Bp + ((size_t)nt0 * KT) * 256 + lane * 8;
    const u32* b1p = b0p + (size_t)KT * 256;
    const u16* Abase = A + (size_t)(mg * 64 + ln) * K;

#pragma unroll 2
    for (int kt = 0; kt < KT; ++kt) {
        FragU b0, b1;
        const uint4* q0 = (const uint4*)(b0p + (size_t)kt * 256);
        const uint4* q1 = (const uint4*)(b1p + (size_t)kt * 256);
        b0.q[0] = q0[0]; b0.q[1] = q0[1];
        b1.q[0] = q1[0]; b1.q[1] = q1[1];
#pragma unroll
        for (int m = 0; m < 4; ++m) {
            FragU a;
            const uint4* ap = (const uint4*)(Abase + (size_t)m * 16 * K + kt * 32);
            a.q[0] = ap[g];
            a.q[1] = ap[2 + g];
            acc[m][0] = __builtin_amdgcn_wmma_f32_16x16x32_bf16(false, a.v, false, b0.v,
                                                               (short)0, acc[m][0], false, false);
            acc[m][1] = __builtin_amdgcn_wmma_f32_16x16x32_bf16(false, a.v, false, b1.v,
                                                               (short)0, acc[m][1], false, false);
        }
    }
    float* Cb = C + (size_t)(mg * 64) * N + nt0 * 16 + ln;
#pragma unroll
    for (int m = 0; m < 4; ++m)
#pragma unroll
        for (int r = 0; r < 8; ++r) {
            Cb[(size_t)(m * 16 + r + 8 * g) * N]      = acc[m][0][r];
            Cb[(size_t)(m * 16 + r + 8 * g) * N + 16] = acc[m][1][r];
        }
}

// ---------- grid barrier across 16 persistent workgroups ----------
__device__ __forceinline__ void grid_barrier(u32* cnt, u32 k) {
    __syncthreads();
    if (threadIdx.x == 0) {
        __threadfence();
        __hip_atomic_fetch_add(cnt, 1u, __ATOMIC_ACQ_REL, __HIP_MEMORY_SCOPE_AGENT);
        while (__hip_atomic_load(cnt, __ATOMIC_ACQUIRE, __HIP_MEMORY_SCOPE_AGENT) < k * 16u) {
            __builtin_amdgcn_s_sleep(2);
        }
    }
    __syncthreads();
}

#if USE_TDM
// TDM: 3D tile load of the per-WG G0x slab for one timestep into LDS.
//   x: 32 contiguous f32 (this WG's 32 gate columns)    -> tile_dim0 = 32
//   y: 4 gate parts, stride 512 f32                     -> tile_dim1 = 4, dim0_stride = 512
//   z: 32 batch rows, stride 2048 f32                   -> tile_dim2 = 32, dim1_stride = 2048
// LDS receives [row][gate][col] f32 = 16 KB, matching the consumer layout.
// Toolchain here exposes the 6-arg builtin: (v4u g0, v8i g1, v4i g2, v4i g3, v8i g4, i32 cpol);
// g0..g3 are the ISA D# groups 0-3, the 5th group has no ISA counterpart (VADDR4 unused) -> zeros.
__device__ __forceinline__ void tdm_load_g0x(const float* src, u32 lds_off) {
    u64 ga = (u64)(uintptr_t)src;
    v4u g0 = { 1u,                                // count=1, is_restore=0, gather off
               lds_off,                           // LDS byte address
               (u32)ga,                           // global_addr[31:0]
               (u32)(ga >> 32) | 0x80000000u };   // global_addr[56:32] | type=2
    v8i g1 = { 0x00020000,                        // data_size=4B, no mask/pad/iterate
               0x00200000,                        // tensor_dim0 = 32 (bits 79:64 -> [31:16])
               0x00040000,                        // tensor_dim1 = 4
               0x00200000,                        // tile_dim0 = 32
               0x00200004,                        // tile_dim1 = 4 | tile_dim2 = 32 << 16
               512,                               // tensor_dim0_stride = 512
               0x08000000,                        // tensor_dim1_stride = 2048 (<<16)
               0 };
    v4i g2 = { 32, 0, 0, 0 };                     // tensor_dim2 = 32
    v4i g3 = { 0, 0, 0, 0 };
    v8i g4 = { 0, 0, 0, 0, 0, 0, 0, 0 };
    __builtin_amdgcn_tensor_load_to_lds(g0, g1, g2, g3, g4, 0);
}
#endif

// ---------- persistent 2-layer LSTM over T steps ----------
// 16 WGs x 512 threads. WG w owns hidden columns [w*32, w*32+32).
// Wave -> one 16x16 gate tile: mt = wave>>3, gate gp = (wave>>1)&3, sub = wave&1.
__global__ __launch_bounds__(512) void lstm_kernel(const float* __restrict__ G0x,
                                                   const u32* __restrict__ Wh0p,
                                                   const u32* __restrict__ Wx1p,
                                                   const u32* __restrict__ Wh1p,
                                                   const float* __restrict__ b1,
                                                   u16* __restrict__ h0d, u16* __restrict__ h1d,
                                                   u16* __restrict__ H1, u32* cnt) {
    const int w = blockIdx.x;
    const int tid = threadIdx.x;
    const int wave = tid >> 5, lane = tid & 31;
    const int g = lane >> 4, ln = lane & 15;
    const int mt = wave >> 3;
    const int gp = (wave >> 1) & 3;
    const int sub = wave & 1;
    const int ntile = gp * 32 + w * 2 + sub;   // N-tile within 2048-wide gates
    const int colbase = ntile << 4;
    __shared__ float glds[32][4][32];
#if USE_TDM
    __shared__ float cin[2][32][4][32];        // double-buffered TDM staging
#endif

    // zero both h double-buffers (this WG's column slice)
    for (int p = tid; p < 2 * 32 * 32; p += 512) {
        int buf = (p >= 1024) ? 1 : 0;
        int q = p & 1023;
        int row = q >> 5, cl = q & 31;
        h0d[buf * 16384 + row * 512 + w * 32 + cl] = 0;
        h1d[buf * 16384 + row * 512 + w * 32 + cl] = 0;
    }
#if USE_TDM
    if (wave == 0)   // prefetch step 0 (G0x already produced by prior kernel on this stream)
        tdm_load_g0x(G0x + (size_t)w * 32, (u32)(uintptr_t)(void*)&cin[0][0][0][0]);
#endif
    // elementwise ownership: elems tid and tid+512 of this WG's 32x32 slice
    const int e0row = tid >> 5,          e0c = tid & 31;
    const int e1row = (tid + 512) >> 5,  e1c = tid & 31;
    float c0a = 0.f, c0b = 0.f, c1a = 0.f, c1b = 0.f;
    u32 bar = 0;
    grid_barrier(cnt, ++bar);

    for (int t = 0; t < TSZ; ++t) {
        const int cur = t & 1, nxt = cur ^ 1;
        const u16* h0c = h0d + cur * 16384;
        u16*       h0n = h0d + nxt * 16384;
        const u16* h1c = h1d + cur * 16384;
        u16*       h1n = h1d + nxt * 16384;

        // ----- layer 0: gates = G0x[t] (has b0) + h0 @ Wh0 -----
        v8f acc;
#if USE_TDM
        if (wave == 0) __builtin_amdgcn_s_wait_tensorcnt(0);
        __syncthreads();                                    // cin[cur] ready for all waves
        if (wave == 0 && t + 1 < TSZ)                       // prefetch next step into cin[nxt]
            tdm_load_g0x(G0x + (size_t)(t + 1) * 32 * G4 + w * 32,
                         (u32)(uintptr_t)(void*)&cin[nxt][0][0][0]);
#pragma unroll
        for (int r = 0; r < 8; ++r) acc[r] = cin[cur][mt * 16 + r + 8 * g][gp][sub * 16 + ln];
#else
        {
            const float* Cin = G0x + ((size_t)(t * 32 + mt * 16)) * G4 + colbase + ln;
#pragma unroll
            for (int r = 0; r < 8; ++r) acc[r] = Cin[(size_t)(r + 8 * g) * G4];
        }
#endif
        acc = tile_mm(h0c + mt * 16 * 512, 512, Wh0p, 16, ntile, acc, lane);
#pragma unroll
        for (int r = 0; r < 8; ++r) glds[mt * 16 + r + 8 * g][gp][sub * 16 + ln] = acc[r];
        __syncthreads();
        {
            float i0 = glds[e0row][0][e0c], j0 = glds[e0row][1][e0c];
            float f0 = glds[e0row][2][e0c], o0 = glds[e0row][3][e0c];
            c0a = sigmoidf_(f0 + 1.f) * c0a + sigmoidf_(i0) * tanhf(j0);
            h0n[e0row * 512 + w * 32 + e0c] = f2bf(sigmoidf_(o0) * tanhf(c0a));
            float i1 = glds[e1row][0][e1c], j1 = glds[e1row][1][e1c];
            float f1 = glds[e1row][2][e1c], o1 = glds[e1row][3][e1c];
            c0b = sigmoidf_(f1 + 1.f) * c0b + sigmoidf_(i1) * tanhf(j1);
            h0n[e1row * 512 + w * 32 + e1c] = f2bf(sigmoidf_(o1) * tanhf(c0b));
        }
        grid_barrier(cnt, ++bar);

        // ----- layer 1: gates = b1 + h0_new @ Wx1 + h1 @ Wh1 -----
        {
            float bv = b1[colbase + ln];
#pragma unroll
            for (int r = 0; r < 8; ++r) acc[r] = bv;
        }
        acc = tile_mm(h0n + mt * 16 * 512, 512, Wx1p, 16, ntile, acc, lane);
        acc = tile_mm(h1c + mt * 16 * 512, 512, Wh1p, 16, ntile, acc, lane);
#pragma unroll
        for (int r = 0; r < 8; ++r) glds[mt * 16 + r + 8 * g][gp][sub * 16 + ln] = acc[r];
        __syncthreads();
        {
            float i0 = glds[e0row][0][e0c], j0 = glds[e0row][1][e0c];
            float f0 = glds[e0row][2][e0c], o0 = glds[e0row][3][e0c];
            c1a = sigmoidf_(f0 + 1.f) * c1a + sigmoidf_(i0) * tanhf(j0);
            u16 hv = f2bf(sigmoidf_(o0) * tanhf(c1a));
            h1n[e0row * 512 + w * 32 + e0c] = hv;
            H1[((size_t)e0row * TSZ + t) * 512 + w * 32 + e0c] = hv;   // batch-major rows
            float i1 = glds[e1row][0][e1c], j1 = glds[e1row][1][e1c];
            float f1 = glds[e1row][2][e1c], o1 = glds[e1row][3][e1c];
            c1b = sigmoidf_(f1 + 1.f) * c1b + sigmoidf_(i1) * tanhf(j1);
            u16 hw = f2bf(sigmoidf_(o1) * tanhf(c1b));
            h1n[e1row * 512 + w * 32 + e1c] = hw;
            H1[((size_t)e1row * TSZ + t) * 512 + w * 32 + e1c] = hw;
        }
        grid_barrier(cnt, ++bar);
    }
}

// ---------- per-row log-softmax NLL, accumulate sum ----------
__global__ __launch_bounds__(256) void softmax_nll_kernel(const float* __restrict__ logits,
                                                          const int* __restrict__ labels,
                                                          float* __restrict__ nll) {
    const int row = blockIdx.x;
    const float* p = logits + (size_t)row * VSZ;
    __shared__ float red[256];
    float m = -3.402823466e38f;
    for (int i = threadIdx.x; i < VSZ; i += 256) m = fmaxf(m, p[i]);
    red[threadIdx.x] = m;
    __syncthreads();
    for (int s = 128; s > 0; s >>= 1) {
        if (threadIdx.x < s) red[threadIdx.x] = fmaxf(red[threadIdx.x], red[threadIdx.x + s]);
        __syncthreads();
    }
    m = red[0];
    __syncthreads();
    float sum = 0.f;
    for (int i = threadIdx.x; i < VSZ; i += 256) sum += __expf(p[i] - m);
    red[threadIdx.x] = sum;
    __syncthreads();
    for (int s = 128; s > 0; s >>= 1) {
        if (threadIdx.x < s) red[threadIdx.x] += red[threadIdx.x + s];
        __syncthreads();
    }
    if (threadIdx.x == 0) {
        float lse = __logf(red[0]);
        float v = m + lse - p[labels[row]];
        atomicAdd(nll, v);
    }
}

__global__ void finalize_kernel(const float* __restrict__ nll, float* __restrict__ out) {
    out[(size_t)BT * VSZ] = __expf(nll[0] * (1.0f / (float)BT));
}

extern "C" void kernel_launch(void* const* d_in, const int* in_sizes, int n_in,
                              void* d_out, int out_size, void* d_ws, size_t ws_size,
                              hipStream_t stream) {
    const int*   inputs = (const int*)d_in[0];
    const int*   labels = (const int*)d_in[1];
    const float* E_mat  = (const float*)d_in[2];
    const float* Wx0    = (const float*)d_in[3];
    const float* Wh0    = (const float*)d_in[4];
    const float* b0     = (const float*)d_in[5];
    const float* Wx1    = (const float*)d_in[6];
    const float* Wh1    = (const float*)d_in[7];
    const float* b1     = (const float*)d_in[8];
    const float* Wd     = (const float*)d_in[9];
    const float* bd     = (const float*)d_in[10];

    char* ws = (char*)d_ws;
    u16*   X    = (u16*)(ws + OFF_X);
    float* G0x  = (float*)(ws + OFF_G0X);
    u32*   Wx0p = (u32*)(ws + OFF_WX0P);
    u32*   Wh0p = (u32*)(ws + OFF_WH0P);
    u32*   Wx1p = (u32*)(ws + OFF_WX1P);
    u32*   Wh1p = (u32*)(ws + OFF_WH1P);
    u32*   Wdp  = (u32*)(ws + OFF_WDP);
    u16*   H1   = (u16*)(ws + OFF_H1);
    u16*   h0d  = (u16*)(ws + OFF_H0S);
    u16*   h1d  = (u16*)(ws + OFF_H1S);
    u32*   cnt  = (u32*)(ws + OFF_CNT);
    float* nll  = (float*)(ws + OFF_NLL);
    float* logits = (float*)d_out;

    (void)hipMemsetAsync(ws + OFF_CNT, 0, 512, stream);   // barrier counter + nll accumulator

    embed_kernel<<<16384, 256, 0, stream>>>(inputs, E_mat, X);

    pack_b_kernel<<<2048, 256, 0, stream>>>(Wx0, Wx0p, ESZ, G4);
    pack_b_kernel<<<2048, 256, 0, stream>>>(Wh0, Wh0p, ESZ, G4);
    pack_b_kernel<<<2048, 256, 0, stream>>>(Wx1, Wx1p, ESZ, G4);
    pack_b_kernel<<<2048, 256, 0, stream>>>(Wh1, Wh1p, ESZ, G4);
    pack_b_kernel<<<16000, 256, 0, stream>>>(Wd, Wdp, ESZ, VSZ);

    // G0x = X @ Wx0 + b0  (time-major rows); groups = (8192/64)*(2048/32) = 8192 -> 1024 blocks
    gemm_bf16_kernel<<<1024, 256, 0, stream>>>(X, Wx0p, b0, G0x, BT, G4, ESZ);

    lstm_kernel<<<16, 512, 0, stream>>>(G0x, Wh0p, Wx1p, Wh1p, b1, h0d, h1d, H1, cnt);

    // logits = H1 @ Wd + bd; groups = (8192/64)*(16000/32) = 64000 -> 8000 blocks
    gemm_bf16_kernel<<<8000, 256, 0, stream>>>(H1, Wdp, bd, logits, BT, VSZ, ESZ);

    softmax_nll_kernel<<<BT, 256, 0, stream>>>(logits, labels, nll);
    finalize_kernel<<<1, 1, 0, stream>>>(nll, logits);
}
